// MultiHeadAttention_40879498723356
// MI455X (gfx1250) — matrix operand order repaired and checked by reading, not compiled, not run on hardware
//
#include <hip/hip_runtime.h>

// ---------- types ----------
typedef __bf16  v16bf __attribute__((ext_vector_type(16)));
typedef float   v8f   __attribute__((ext_vector_type(8)));
typedef unsigned short ushort_t;

union FragBF {
    v16bf v;
    uint4 q[2];
};

static __device__ __forceinline__ ushort_t f32_to_bf16(float f) {
    unsigned int u = __float_as_uint(f);
    unsigned int r = (u + 0x7FFFu + ((u >> 16) & 1u)) >> 16;
    return (ushort_t)r;
}

// ---------- CDNA5 async global->LDS (ASYNCcnt path) ----------
static __device__ __forceinline__ void async_load_b128(const ushort_t* g,
                                                       ushort_t* lds) {
    // VDST = LDS byte address (low 32 bits of generic LDS pointer),
    // VADDR = 64-bit global address.
    unsigned l = (unsigned)(size_t)lds;
    asm volatile("global_load_async_to_lds_b128 %0, %1, off"
                 :
                 : "v"(l), "v"(g)
                 : "memory");
}
static __device__ __forceinline__ void wait_async0() {
    asm volatile("s_wait_asynccnt 0" ::: "memory");
}

#define LOG2E 1.44269504088896f

// ---------- fp32 -> bf16 convert (vectorized x4) ----------
__global__ __launch_bounds__(256) void cvt_bf16_kernel(const float* __restrict__ in,
                                                       ushort_t* __restrict__ out, int n) {
    int i = (blockIdx.x * blockDim.x + threadIdx.x) * 4;
    if (i + 3 < n) {
        float4 f = *(const float4*)(in + i);
        out[i + 0] = f32_to_bf16(f.x);
        out[i + 1] = f32_to_bf16(f.y);
        out[i + 2] = f32_to_bf16(f.z);
        out[i + 3] = f32_to_bf16(f.w);
    }
}

// ---------- fp32 -> bf16 convert + transpose: out[c*R + r] = in[r*C + c] ----------
__global__ __launch_bounds__(256) void cvt_bf16_t_kernel(const float* __restrict__ in,
                                                         ushort_t* __restrict__ out,
                                                         int R, int C) {
    int idx = blockIdx.x * blockDim.x + threadIdx.x;
    if (idx < R * C) {
        int r = idx / C;
        int c = idx - r * C;
        out[(size_t)c * R + r] = f32_to_bf16(in[idx]);
    }
}

// ---------- bf16 GEMM: C[M,N] = A[M,K] * Bt[N,K]^T, block tile 128x64 ----------
// A row-major [M,Kd]; Bt row-major [N,Kd] (pre-transposed weight).
// EPI == 0 : write fp32 C row-major
// EPI == 1 : scatter into Q/K [B,H,T,DH] and V-transposed [B,H,DH,T]; Q scaled 0.125
template <int EPI>
__global__ __launch_bounds__(256, 2) void gemm_bf16_kernel(
    const ushort_t* __restrict__ A, const ushort_t* __restrict__ Bt,
    float* __restrict__ Cout, ushort_t* __restrict__ Qb, ushort_t* __restrict__ Kb,
    ushort_t* __restrict__ Vt, int M, int N, int Kd) {
    __shared__ ushort_t As[2][128][40];  // double buffered, row stride 80B
    __shared__ ushort_t Bs[2][64][40];   // Bs[n][k]

    const int tid   = threadIdx.x;
    const int lane  = tid & 31;
    const int wave  = tid >> 5;
    const int waveM = wave >> 1;  // 0..3
    const int waveN = wave & 1;   // 0..1
    const int l15   = lane & 15;
    const int lhi   = lane >> 4;  // 0 or 1
    const int hi8   = lhi * 8;
    const int m0    = blockIdx.y * 128;
    const int n0    = blockIdx.x * 64;

    auto issueTile = [&](int kk, int buf) {
        // A tile 128x32: 512 x 16B chunks, 2 per thread
#pragma unroll
        for (int it = 0; it < 2; ++it) {
            int cid = tid + it * 256;
            int row = cid >> 2;
            int c8  = (cid & 3) * 8;
            async_load_b128(A + (size_t)(m0 + row) * Kd + kk + c8,
                            &As[buf][row][c8]);
        }
        // B tile 64x32 from Bt rows: 256 x 16B chunks, 1 per thread
        {
            int row = tid >> 2;
            int c8  = (tid & 3) * 8;
            async_load_b128(Bt + (size_t)(n0 + row) * Kd + kk + c8,
                            &Bs[buf][row][c8]);
        }
    };

    v8f acc[2][2] = {};

    issueTile(0, 0);
    wait_async0();
    __syncthreads();
    int cur = 0;

    for (int kk = 0; kk < Kd; kk += 32) {
        if (kk + 32 < Kd) issueTile(kk + 32, cur ^ 1);

        FragBF af[2], bf[2];
#pragma unroll
        for (int tm = 0; tm < 2; ++tm) {
            const ushort_t* ar = &As[cur][waveM * 32 + tm * 16 + l15][lhi * 8];
            af[tm].q[0] = *(const uint4*)(ar);
            af[tm].q[1] = *(const uint4*)(ar + 16);
        }
#pragma unroll
        for (int tn = 0; tn < 2; ++tn) {
            const ushort_t* br = &Bs[cur][waveN * 32 + tn * 16 + l15][lhi * 8];
            bf[tn].q[0] = *(const uint4*)(br);
            bf[tn].q[1] = *(const uint4*)(br + 16);
        }
#pragma unroll
        for (int tm = 0; tm < 2; ++tm)
#pragma unroll
            for (int tn = 0; tn < 2; ++tn)
                acc[tm][tn] = __builtin_amdgcn_wmma_f32_16x16x32_bf16(
                    false, af[tm].v, false, bf[tn].v, (short)0, acc[tm][tn],
                    false, false);

        wait_async0();
        __syncthreads();
        cur ^= 1;
    }

    // epilogue: C element (r, lane) -> row = r + 8*lhi, col = l15
#pragma unroll
    for (int tm = 0; tm < 2; ++tm) {
#pragma unroll
        for (int tn = 0; tn < 2; ++tn) {
#pragma unroll
            for (int r = 0; r < 8; ++r) {
                int gm = m0 + waveM * 32 + tm * 16 + r + hi8;
                int gn = n0 + waveN * 32 + tn * 16 + l15;
                float v = acc[tm][tn][r];
                if (EPI == 0) {
                    Cout[(size_t)gm * N + gn] = v;
                } else {
                    // col factorization: gn = d*48 + kq*16 + h
                    int bb = gm >> 11;  // T = 2048
                    int t  = gm & 2047;
                    int hh = gn & 15;
                    int kq = (gn >> 4) % 3;
                    int d  = gn / 48;
                    if (kq == 2) {
                        // V transposed: [B,H,DH,T]
                        Vt[(((size_t)bb * 16 + hh) * 64 + d) * 2048 + t] =
                            f32_to_bf16(v);
                    } else {
                        ushort_t* dst = (kq == 0) ? Qb : Kb;
                        float sv = (kq == 0) ? v * 0.125f : v;  // DH^-0.5 on Q
                        dst[(((size_t)bb * 16 + hh) * 2048 + t) * 64 + d] =
                            f32_to_bf16(sv);
                    }
                }
            }
        }
    }
}

// ---------- flash attention: Q/K [B,H,T,DH] bf16, V [B,H,DH,T] bf16 ----------
// grid.x = (B*H) * (T/128) = 512 ; 256 threads (8 waves x 16 Q rows)
__global__ __launch_bounds__(256, 2) void attn_kernel(
    const ushort_t* __restrict__ Qb, const ushort_t* __restrict__ Kb,
    const ushort_t* __restrict__ Vt, ushort_t* __restrict__ AO) {
    __shared__ ushort_t Ks[2][64][80];   // [key][d], double buffered
    __shared__ ushort_t Vs[2][64][80];   // [d][key], double buffered
    __shared__ ushort_t Ps[8][16][80];   // per-wave probs [m][key]

    const int tid  = threadIdx.x;
    const int lane = tid & 31;
    const int wave = tid >> 5;
    const int bh   = blockIdx.x >> 4;  // 0..31
    const int rblk = blockIdx.x & 15;  // 0..15
    const int b    = bh >> 4;
    const int h    = bh & 15;
    const int t0   = rblk * 128 + wave * 16;
    const int l15  = lane & 15;
    const int lhi  = lane >> 4;
    const int hi8  = lhi * 8;

    auto issueKV = [&](int j0, int buf) {
#pragma unroll
        for (int it = 0; it < 2; ++it) {
            int cid = tid + it * 256;
            int row = cid >> 3;       // 0..63
            int c8  = (cid & 7) * 8;  // chunk
            async_load_b128(Kb + ((size_t)bh * 2048 + j0 + row) * 64 + c8,
                            &Ks[buf][row][c8]);
            async_load_b128(Vt + ((size_t)bh * 64 + row) * 2048 + j0 + c8,
                            &Vs[buf][row][c8]);
        }
    };

    // Q fragments for this wave's 16 rows (A-matrix layout), held in regs
    const ushort_t* qrow = Qb + ((size_t)bh * 2048 + (t0 + l15)) * 64;
    const int kb = lhi * 8;
    FragBF qf[2];
    qf[0].q[0] = *(const uint4*)(qrow + kb);
    qf[0].q[1] = *(const uint4*)(qrow + kb + 16);
    qf[1].q[0] = *(const uint4*)(qrow + kb + 32);
    qf[1].q[1] = *(const uint4*)(qrow + kb + 48);

    // constant all-ones B fragment (1.0 bf16 = 0x3F80): rowsum(P) = P @ ones
    FragBF ones;
    ones.q[0] = make_uint4(0x3F803F80u, 0x3F803F80u, 0x3F803F80u, 0x3F803F80u);
    ones.q[1] = make_uint4(0x3F803F80u, 0x3F803F80u, 0x3F803F80u, 0x3F803F80u);

    float mrow[8], lrow[8];
#pragma unroll
    for (int r = 0; r < 8; ++r) { mrow[r] = -1e30f; lrow[r] = 0.f; }
    v8f oacc[4] = {};

    issueKV(0, 0);
    wait_async0();
    __syncthreads();
    int cur = 0;

    for (int j0 = 0; j0 < 2048; j0 += 64) {
        if (j0 + 64 < 2048) issueKV(j0 + 64, cur ^ 1);

        // S = Q * K^T  (16 x 64 per wave)
        v8f sacc[4] = {};
#pragma unroll
        for (int ks = 0; ks < 2; ++ks) {
#pragma unroll
            for (int sn = 0; sn < 4; ++sn) {
                FragBF bk;
                const ushort_t* kr = &Ks[cur][sn * 16 + l15][lhi * 8 + ks * 32];
                bk.q[0] = *(const uint4*)(kr);
                bk.q[1] = *(const uint4*)(kr + 16);
                sacc[sn] = __builtin_amdgcn_wmma_f32_16x16x32_bf16(
                    false, qf[ks].v, false, bk.v, (short)0, sacc[sn], false,
                    false);
            }
        }

        // online softmax: row-max via xor-shuffles (rows live in 16-lane halves),
        // exp + rescale; row-sum deferred to a P @ ones WMMA below.
#pragma unroll
        for (int r = 0; r < 8; ++r) {
            float v = fmaxf(fmaxf(sacc[0][r], sacc[1][r]),
                            fmaxf(sacc[2][r], sacc[3][r]));
#pragma unroll
            for (int off = 1; off < 16; off <<= 1)
                v = fmaxf(v, __shfl_xor(v, off, 32));
            float mnew = fmaxf(mrow[r], v);
            float corr = exp2f((mrow[r] - mnew) * LOG2E);
            mrow[r] = mnew;
            lrow[r] *= corr;
#pragma unroll
            for (int s = 0; s < 4; ++s) {
                float p = exp2f((sacc[s][r] - mnew) * LOG2E);
                Ps[wave][r + hi8][s * 16 + l15] = f32_to_bf16(p);
                oacc[s][r] *= corr;
            }
        }

        // O += P * V ; lacc = P @ ones gives per-row sums on the matrix pipe
        v8f lacc = {};
#pragma unroll
        for (int ks = 0; ks < 2; ++ks) {
            FragBF pf;
            const ushort_t* pr = &Ps[wave][l15][lhi * 8 + ks * 32];
            pf.q[0] = *(const uint4*)(pr);
            pf.q[1] = *(const uint4*)(pr + 16);
            lacc = __builtin_amdgcn_wmma_f32_16x16x32_bf16(
                false, pf.v, false, ones.v, (short)0, lacc, false, false);
#pragma unroll
            for (int s = 0; s < 4; ++s) {
                FragBF bv;
                const ushort_t* vr = &Vs[cur][s * 16 + l15][lhi * 8 + ks * 32];
                bv.q[0] = *(const uint4*)(vr);
                bv.q[1] = *(const uint4*)(vr + 16);
                oacc[s] = __builtin_amdgcn_wmma_f32_16x16x32_bf16(
                    false, pf.v, false, bv.v, (short)0, oacc[s], false, false);
            }
        }
#pragma unroll
        for (int r = 0; r < 8; ++r) lrow[r] += lacc[r];

        wait_async0();
        __syncthreads();
        cur ^= 1;
    }

    // normalize and write [B*T, H*DH] bf16 for the output projection
#pragma unroll
    for (int r = 0; r < 8; ++r) {
        float inv = 1.0f / lrow[r];
        int t = t0 + r + hi8;
        size_t base = ((size_t)b * 2048 + t) * 1024 + h * 64;
#pragma unroll
        for (int s = 0; s < 4; ++s)
            AO[base + s * 16 + l15] = f32_to_bf16(oacc[s][r] * inv);
    }
}

// ---------- launcher ----------
extern "C" void kernel_launch(void* const* d_in, const int* in_sizes, int n_in,
                              void* d_out, int out_size, void* d_ws,
                              size_t ws_size, hipStream_t stream) {
    const float* x    = (const float*)d_in[0];  // [2,2048,1024]
    const float* wqkv = (const float*)d_in[1];  // [1024,3072]
    const float* wout = (const float*)d_in[2];  // [1024,1024]
    float* out        = (float*)d_out;          // [2,2048,1024]

    const size_t XN = (size_t)4096 * 1024;
    const size_t WQ = (size_t)1024 * 3072;
    const size_t WO = (size_t)1024 * 1024;
    const size_t QN = (size_t)32 * 2048 * 64;  // B*H*T*DH

    ushort_t* ws       = (ushort_t*)d_ws;
    ushort_t* x_bf     = ws;
    ushort_t* wqkvT_bf = x_bf + XN;      // [3072,1024]
    ushort_t* woutT_bf = wqkvT_bf + WQ;  // [1024,1024]
    ushort_t* Qb       = woutT_bf + WO;  // [B,H,T,DH]
    ushort_t* Kb       = Qb + QN;        // [B,H,T,DH]
    ushort_t* Vt       = Kb + QN;        // [B,H,DH,T]
    ushort_t* AO       = Vt + QN;        // [B*T, H*DH]

    // converts (x plain; weights transposed for async-friendly B tiles)
    cvt_bf16_kernel<<<(int)(XN / 1024), 256, 0, stream>>>(x, x_bf, (int)XN);
    cvt_bf16_t_kernel<<<(int)((WQ + 255) / 256), 256, 0, stream>>>(wqkv, wqkvT_bf,
                                                                   1024, 3072);
    cvt_bf16_t_kernel<<<(int)((WO + 255) / 256), 256, 0, stream>>>(wout, woutT_bf,
                                                                   1024, 1024);

    // qkv = x @ w_qkv, scattered into Q/K (scaled Q) and transposed V
    dim3 g1(3072 / 64, 4096 / 128);
    gemm_bf16_kernel<1><<<g1, 256, 0, stream>>>(x_bf, wqkvT_bf, nullptr, Qb, Kb,
                                                Vt, 4096, 3072, 1024);

    // attention
    attn_kernel<<<512, 256, 0, stream>>>(Qb, Kb, Vt, AO);

    // out = attn_out @ w_out (fp32)
    dim3 g2(1024 / 64, 4096 / 128);
    gemm_bf16_kernel<0><<<g2, 256, 0, stream>>>(AO, woutT_bf, out, nullptr,
                                                nullptr, nullptr, 4096, 1024,
                                                1024);
}
